// DecoderLayer_33406255628759
// MI455X (gfx1250) — matrix operand order repaired and checked by reading, not compiled, not run on hardware
//
#include <hip/hip_runtime.h>
#include <hip/hip_bf16.h>

// ---------------------------------------------------------------------------
// DecoderLayer for MI455X (gfx1250): bf16 WMMA GEMMs + flash attention.
// B=4, T=2048, D=1024, H=16, HD=64, F=4096
// GEMM staging: A tile via async global->LDS DMA (ASYNCcnt), B tile via the
// Tensor Data Mover (TENSORcnt) with hardware LDS padding; double buffered.
// ---------------------------------------------------------------------------

#define DEV static __device__ __forceinline__

typedef unsigned short u16;
typedef __attribute__((ext_vector_type(16))) __bf16 bf16x16;
typedef __attribute__((ext_vector_type(8)))  float  f32x8;
typedef __attribute__((ext_vector_type(4)))  unsigned int v4u;
typedef __attribute__((ext_vector_type(8)))  int v8i;
typedef __attribute__((ext_vector_type(4)))  int v4i;

static_assert(sizeof(bf16x16) == 32, "bf16x16 must be 32B");

struct alignas(16) U128 { unsigned int v[4]; };
struct U256 { U128 lo, hi; };

DEV u16 f2bf(float f) {
    unsigned int u = __float_as_uint(f);
    u += 0x7fffu + ((u >> 16) & 1u);          // round-to-nearest-even
    return (u16)(u >> 16);
}

DEV bf16x16 make_frag(U128 lo, U128 hi) {
    U256 t{lo, hi};
    return __builtin_bit_cast(bf16x16, t);
}

#define WMMA_BF16(a, b, c) \
    __builtin_amdgcn_wmma_f32_16x16x32_bf16(false, (a), false, (b), (short)0, (c), false, false)

// Async global->LDS b128 (ASYNCcnt-tracked).
DEV void async_ld_b128(unsigned ldsoff, unsigned long long gaddr) {
    asm volatile("global_load_async_to_lds_b128 %0, %1, off"
                 :: "v"(ldsoff), "v"(gaddr) : "memory");
}
DEV void wait_async0() {
    asm volatile("s_wait_asynccnt 0x0" ::: "memory");
}

// Tensor Data Mover: 2D tile load, 128 rows x 32 bf16 elements, row stride
// ldK elements, with HW LDS padding: 4 DWORDs pad after every 16 DWORDs
// (-> 80B LDS row pitch == our stride-40-ushort layout).
DEV void tdm_load_tile(unsigned ldsaddr, const u16* g, int ldK) {
    unsigned long long ga = (unsigned long long)(const void*)g;
    v4u g0 = { 1u,                                  // count=1 (valid descriptor)
               ldsaddr,                             // lds_addr (bytes)
               (unsigned)ga,                        // global_addr[31:0]
               (unsigned)((ga >> 32) & 0x1FFFFFFull) | 0x80000000u };  // addr[56:32] | type=2
    unsigned ldKu = (unsigned)ldK;
    v8i g1 = {
        (int)((1u << 16) | (1u << 20) | (3u << 22) | (3u << 25)),
        //      data_size=2B  pad_en     pad_int=16DW  pad_amt=4DW ; wg_mask=0
        (int)((ldKu & 0xFFFFu) << 16),              // [15:0] atomic_bar=0, [31:16] tensor_dim0.lo
        (int)(((ldKu >> 16) & 0xFFFFu) | (128u << 16)),  // tensor_dim0.hi | tensor_dim1.lo=128
        (int)(0u | (32u << 16)),                    // tensor_dim1.hi=0 | tile_dim0=32
        (int)(128u),                                // tile_dim1=128 | tile_dim2=0
        (int)ldKu,                                  // tensor_dim0_stride[31:0] (elements)
        0,                                          // stride.hi | dim1_stride.lo
        0 };
    v4i z4 = {0, 0, 0, 0};
#if __clang_major__ >= 23
    v8i z8 = {0, 0, 0, 0, 0, 0, 0, 0};
    __builtin_amdgcn_tensor_load_to_lds(g0, g1, z4, z4, z8, 0);
#else
    __builtin_amdgcn_tensor_load_to_lds(g0, g1, z4, z4, 0);
#endif
}

// ---------------------------------------------------------------------------
// Weight casts
// ---------------------------------------------------------------------------

// Wq/Wk/Wv (H,D,HD) f32 -> Wqkv_t bf16 [3][N=1024][K=1024], N = h*64+j, K = d
__global__ __launch_bounds__(256) void cast_qkv_kernel(
        const float* __restrict__ Wq, const float* __restrict__ Wk,
        const float* __restrict__ Wv, u16* __restrict__ out) {
    unsigned idx = blockIdx.x * 256u + threadIdx.x;     // [0, 3*2^20)
    unsigned s   = idx >> 20;
    unsigned rem = idx & 0xFFFFFu;
    unsigned n   = rem >> 10;                           // output col (h*64+j)
    unsigned kk  = rem & 1023u;                         // k = d
    unsigned h   = n >> 6, j = n & 63u;
    const float* W = (s == 0) ? Wq : (s == 1) ? Wk : Wv;
    out[idx] = f2bf(W[((size_t)h * 1024u + kk) * 64u + j]);   // h*D*HD + d*HD + j
}

// in [K][N] f32 row-major -> out [N][K] bf16
__global__ __launch_bounds__(256) void cast_T_kernel(
        const float* __restrict__ in, u16* __restrict__ out, int N, int K) {
    unsigned idx = blockIdx.x * 256u + threadIdx.x;
    unsigned n = idx / (unsigned)K, kk = idx % (unsigned)K;
    out[idx] = f2bf(in[(size_t)kk * N + n]);
}

// ---------------------------------------------------------------------------
// LayerNorm: [8192][1024] f32 -> bf16, one row per block
// ---------------------------------------------------------------------------
__global__ __launch_bounds__(256) void ln_kernel(
        const float* __restrict__ in, const float* __restrict__ g,
        const float* __restrict__ b, u16* __restrict__ out) {
    const int row = blockIdx.x, tid = threadIdx.x;
    const float* rp = in + (size_t)row * 1024;
    float x0 = rp[tid * 4 + 0], x1 = rp[tid * 4 + 1];
    float x2 = rp[tid * 4 + 2], x3 = rp[tid * 4 + 3];
    __shared__ float r1[256], r2[256];
    r1[tid] = x0 + x1 + x2 + x3;
    r2[tid] = x0 * x0 + x1 * x1 + x2 * x2 + x3 * x3;
    __syncthreads();
    for (int o = 128; o > 0; o >>= 1) {
        if (tid < o) { r1[tid] += r1[tid + o]; r2[tid] += r2[tid + o]; }
        __syncthreads();
    }
    float mu   = r1[0] * (1.0f / 1024.0f);
    float var  = r2[0] * (1.0f / 1024.0f) - mu * mu;
    float rstd = rsqrtf(var + 1e-5f);
    float xs[4] = {x0, x1, x2, x3};
#pragma unroll
    for (int i = 0; i < 4; ++i) {
        int c = tid * 4 + i;
        out[(size_t)row * 1024 + c] = f2bf((xs[i] - mu) * rstd * g[c] + b[c]);
    }
}

// ---------------------------------------------------------------------------
// Tiled bf16 GEMM: C[M,N] = A[M,K] * Bt[N,K]^T, f32 accumulate via WMMA.
// Block tile 128x128x32, 256 threads = 8 waves (4x2), wave tile 32x64.
// A tile: per-lane async global->LDS.  B tile: one TDM descriptor (wave 0).
// Epilogues: 0=QKV scatter, 1=O-proj(+bias+resid->f32), 2=ReLU(+bias->bf16),
//            3=FFN2 (outF += acc + bias).
// ---------------------------------------------------------------------------
__global__ __launch_bounds__(256) void gemm_bf16_kernel(
        const u16* __restrict__ A, const u16* __restrict__ Bt,
        int M, int N, int K, int epi,
        const float* __restrict__ bias, const float* __restrict__ resid,
        float* __restrict__ outF, u16* __restrict__ outB,
        u16* __restrict__ qp, u16* __restrict__ kp, u16* __restrict__ vp) {
    (void)M;
    __shared__ u16 sA[2][128 * 40];   // stride 40 (=32+8) dodges bank conflicts
    __shared__ u16 sB[2][128 * 40];   // same pitch, produced by TDM HW padding

    const int tid  = threadIdx.x;
    const int lane = tid & 31, w = tid >> 5;
    const int half = lane >> 4, ln = lane & 15;
    const int wm = w >> 1, wn = w & 1;
    const int m0 = blockIdx.y * 128, n0 = blockIdx.x * 128;

    // Per-thread async slice of the 128x32 A tile: 2 chunks of 16B.
    const int c0  = tid, c1 = tid + 256;
    const int r0  = c0 >> 2, o0 = (c0 & 3) * 8;
    const int r1  = c1 >> 2, o1 = (c1 & 3) * 8;

    auto issue_tile = [&](int bufi, int k0) {
        unsigned a0 = (unsigned)(unsigned long long)(const void*)&sA[bufi][r0 * 40 + o0];
        unsigned a1 = (unsigned)(unsigned long long)(const void*)&sA[bufi][r1 * 40 + o1];
        async_ld_b128(a0, (unsigned long long)(const void*)&A[(size_t)(m0 + r0) * K + k0 + o0]);
        async_ld_b128(a1, (unsigned long long)(const void*)&A[(size_t)(m0 + r1) * K + k0 + o1]);
        if (w == 0) {
            unsigned bl = (unsigned)(unsigned long long)(const void*)&sB[bufi][0];
            tdm_load_tile(bl, &Bt[(size_t)n0 * K + k0], K);
        }
    };

    f32x8 zero8 = {0, 0, 0, 0, 0, 0, 0, 0};
    f32x8 acc[2][4];
#pragma unroll
    for (int mt = 0; mt < 2; ++mt)
#pragma unroll
        for (int nt = 0; nt < 4; ++nt) acc[mt][nt] = zero8;

    issue_tile(0, 0);                         // prefetch tile 0
    int buf = 0;
    const int nk = K >> 5;
    for (int kt = 0; kt < nk; ++kt) {
        if (w == 0) __builtin_amdgcn_s_wait_tensorcnt(0);  // B tile landed
        wait_async0();                        // my A slice landed
        __syncthreads();                      // publish LDS to all waves
        if (kt + 1 < nk) issue_tile(buf ^ 1, (kt + 1) * 32);  // overlap DMA

        // A fragment: lane row = ln; K chunks at half*8 and 16+half*8
        bf16x16 af[2];
#pragma unroll
        for (int mt = 0; mt < 2; ++mt) {
            int row = wm * 32 + mt * 16 + ln;
            af[mt] = make_frag(*(const U128*)&sA[buf][row * 40 + half * 8],
                               *(const U128*)&sA[buf][row * 40 + 16 + half * 8]);
        }
        // B fragment: lane col = ln; K chunks at half*8 and 16+half*8
        bf16x16 bfr[4];
#pragma unroll
        for (int nt = 0; nt < 4; ++nt) {
            int row = wn * 64 + nt * 16 + ln;
            bfr[nt] = make_frag(*(const U128*)&sB[buf][row * 40 + half * 8],
                                *(const U128*)&sB[buf][row * 40 + 16 + half * 8]);
        }
#pragma unroll
        for (int mt = 0; mt < 2; ++mt)
#pragma unroll
            for (int nt = 0; nt < 4; ++nt)
                acc[mt][nt] = WMMA_BF16(af[mt], bfr[nt], acc[mt][nt]);
        buf ^= 1;
    }

    // Epilogue. C layout: VGPR r -> row half*8+r, col = ln.
#pragma unroll
    for (int mt = 0; mt < 2; ++mt)
#pragma unroll
        for (int nt = 0; nt < 4; ++nt)
#pragma unroll
            for (int r = 0; r < 8; ++r) {
                float v = acc[mt][nt][r];
                int gm = m0 + wm * 32 + mt * 16 + half * 8 + r;
                int gn = n0 + wn * 64 + nt * 16 + ln;
                if (epi == 0) {               // QKV scatter (bf16)
                    int s = gn >> 10, nn = gn & 1023;
                    int h = nn >> 6, j = nn & 63;
                    int bb = gm >> 11, t = gm & 2047;
                    size_t bh = (size_t)(bb * 16 + h);
                    if (s == 0)      qp[(bh * 2048 + t) * 64 + j] = f2bf(v);
                    else if (s == 1) kp[(bh * 2048 + t) * 64 + j] = f2bf(v);
                    else             vp[(bh * 64 + j) * 2048 + t] = f2bf(v);  // V transposed
                } else if (epi == 1) {        // h = attn@Wo + bo + x  (f32)
                    size_t o = (size_t)gm * 1024 + gn;
                    outF[o] = v + bias[gn] + resid[o];
                } else if (epi == 2) {        // relu(hn@W1 + b1) -> bf16
                    float u = v + bias[gn];
                    outB[(size_t)gm * N + gn] = f2bf(u > 0.0f ? u : 0.0f);
                } else {                      // out = h + ff@W2 + b2 (f32, h in outF)
                    size_t o = (size_t)gm * 1024 + gn;
                    outF[o] = outF[o] + v + bias[gn];
                }
            }
}

// ---------------------------------------------------------------------------
// Flash attention. grid(B*H, T/64), 128 thr = 4 waves; wave owns 16 q-rows.
// q,k: [bh][T][64] bf16 ; vt: [bh][64][T] bf16 ; out concat: [b*T][1024] bf16
// ---------------------------------------------------------------------------
__global__ __launch_bounds__(128) void attn_kernel(
        const u16* __restrict__ q, const u16* __restrict__ k,
        const u16* __restrict__ vt, u16* __restrict__ cat) {
    const int tid = threadIdx.x, lane = tid & 31, w = tid >> 5;
    const int half = lane >> 4, ln = lane & 15;
    const int bh = blockIdx.x, b = bh >> 4, h = bh & 15;
    const int qb = blockIdx.y * 64 + w * 16;
    const size_t off = (size_t)bh * 2048 * 64;
    const u16* qB = q + off;
    const u16* kB = k + off;
    const u16* vB = vt + off;
    __shared__ u16 sP[4][16][40];   // per-wave P staging (C-layout -> A-layout)

    // Q fragments (K halves 0-31, 32-63), straight from global
    bf16x16 aq[2];
#pragma unroll
    for (int kk = 0; kk < 2; ++kk) {
        const u16* rp = &qB[(size_t)(qb + ln) * 64 + kk * 32];
        aq[kk] = make_frag(*(const U128*)&rp[half * 8], *(const U128*)&rp[16 + half * 8]);
    }

    f32x8 zero8 = {0, 0, 0, 0, 0, 0, 0, 0};
    f32x8 O[4] = {zero8, zero8, zero8, zero8};
    float mr[8], lr[8];
#pragma unroll
    for (int r = 0; r < 8; ++r) { mr[r] = -3.0e38f; lr[r] = 0.0f; }

    const float scale = 0.03125f;                 // D^-0.5 = 1024^-0.5
    const float L2E   = 1.4426950408889634f;

    for (int s0 = 0; s0 < qb + 16; s0 += 32) {
        f32x8 S[2] = {zero8, zero8};
#pragma unroll
        for (int st = 0; st < 2; ++st)
#pragma unroll
            for (int kk = 0; kk < 2; ++kk) {
                const u16* rp = &kB[(size_t)(s0 + st * 16 + ln) * 64 + kk * 32];
                bf16x16 bk = make_frag(*(const U128*)&rp[half * 8],
                                       *(const U128*)&rp[16 + half * 8]);
                S[st] = WMMA_BF16(aq[kk], bk, S[st]);
            }
        // streamed softmax (per row = half*8 + r; 16 cols per half-group)
#pragma unroll
        for (int r = 0; r < 8; ++r) {
            int t = qb + half * 8 + r;
            float v0 = S[0][r] * scale, v1 = S[1][r] * scale;
            v0 = (s0 + ln      <= t) ? v0 : -3.0e38f;
            v1 = (s0 + 16 + ln <= t) ? v1 : -3.0e38f;
            float rm = fmaxf(v0, v1);
            rm = fmaxf(rm, __shfl_xor(rm, 1));
            rm = fmaxf(rm, __shfl_xor(rm, 2));
            rm = fmaxf(rm, __shfl_xor(rm, 4));
            rm = fmaxf(rm, __shfl_xor(rm, 8));
            float mnew  = fmaxf(mr[r], rm);
            float alpha = exp2f((mr[r] - mnew) * L2E);
            float p0 = exp2f((v0 - mnew) * L2E);
            float p1 = exp2f((v1 - mnew) * L2E);
            float rs = p0 + p1;
            rs += __shfl_xor(rs, 1);
            rs += __shfl_xor(rs, 2);
            rs += __shfl_xor(rs, 4);
            rs += __shfl_xor(rs, 8);
            lr[r] = lr[r] * alpha + rs;
            mr[r] = mnew;
#pragma unroll
            for (int nt = 0; nt < 4; ++nt) O[nt][r] *= alpha;
            int row = half * 8 + r;
            sP[w][row][ln]      = f2bf(p0);
            sP[w][row][16 + ln] = f2bf(p1);
        }
        asm volatile("s_wait_dscnt 0x0" ::: "memory");   // wave's ds stores done
        bf16x16 ap = make_frag(*(const U128*)&sP[w][ln][half * 8],
                               *(const U128*)&sP[w][ln][16 + half * 8]);
#pragma unroll
        for (int nt = 0; nt < 4; ++nt) {
            const u16* rp = &vB[(size_t)(nt * 16 + ln) * 2048 + s0];
            bf16x16 bv = make_frag(*(const U128*)&rp[half * 8],
                                   *(const U128*)&rp[16 + half * 8]);
            O[nt] = WMMA_BF16(ap, bv, O[nt]);
        }
    }
#pragma unroll
    for (int nt = 0; nt < 4; ++nt)
#pragma unroll
        for (int r = 0; r < 8; ++r) {
            int t = qb + half * 8 + r;
            int col = h * 64 + nt * 16 + ln;
            cat[(size_t)(b * 2048 + t) * 1024 + col] = f2bf(O[nt][r] / lr[r]);
        }
}

// ---------------------------------------------------------------------------
// Host launcher
// ---------------------------------------------------------------------------
extern "C" void kernel_launch(void* const* d_in, const int* in_sizes, int n_in,
                              void* d_out, int out_size, void* d_ws, size_t ws_size,
                              hipStream_t stream) {
    (void)in_sizes; (void)n_in; (void)out_size; (void)ws_size;
    const float* x     = (const float*)d_in[0];
    const float* ln1_g = (const float*)d_in[1];
    const float* ln1_b = (const float*)d_in[2];
    const float* ln2_g = (const float*)d_in[3];
    const float* ln2_b = (const float*)d_in[4];
    const float* Wq    = (const float*)d_in[5];
    const float* Wk    = (const float*)d_in[6];
    const float* Wv    = (const float*)d_in[7];
    const float* Wo    = (const float*)d_in[8];
    const float* bo    = (const float*)d_in[9];
    const float* W1    = (const float*)d_in[10];
    const float* b1    = (const float*)d_in[11];
    const float* W2    = (const float*)d_in[12];
    const float* b2    = (const float*)d_in[13];
    float* out = (float*)d_out;

    const size_t MB = 1u << 20;
    char* ws = (char*)d_ws;
    u16* act  = (u16*)(ws + 0 * MB);     // 16 MB: xn, later hn (bf16 [8192][1024])
    u16* qbuf = (u16*)(ws + 16 * MB);    // 16 MB: [bh][T][64]
    u16* kbuf = (u16*)(ws + 32 * MB);    // 16 MB: [bh][T][64]
    u16* vtb  = (u16*)(ws + 48 * MB);    // 16 MB: [bh][64][T]
    u16* cat  = (u16*)(ws + 64 * MB);    // 16 MB: attn concat bf16 [8192][1024]
    u16* ff1  = (u16*)(ws + 80 * MB);    // 64 MB: relu out bf16 [8192][4096]
    u16* wqkv = (u16*)(ws + 144 * MB);   //  6 MB: [3][1024][1024]
    u16* wot  = (u16*)(ws + 150 * MB);   //  2 MB
    u16* w1t  = (u16*)(ws + 152 * MB);   //  8 MB: [4096][1024]
    u16* w2t  = (u16*)(ws + 160 * MB);   //  8 MB: [1024][4096]

    // 1) weight casts (f32 -> bf16, B-transposed layouts)
    cast_qkv_kernel<<<(3u << 20) / 256, 256, 0, stream>>>(Wq, Wk, Wv, wqkv);
    cast_T_kernel<<<(1024u * 1024) / 256, 256, 0, stream>>>(Wo, wot, 1024, 1024);
    cast_T_kernel<<<(4096u * 1024) / 256, 256, 0, stream>>>(W1, w1t, 4096, 1024);
    cast_T_kernel<<<(4096u * 1024) / 256, 256, 0, stream>>>(W2, w2t, 1024, 4096);

    // 2) LN1 -> xn (bf16)
    ln_kernel<<<8192, 256, 0, stream>>>(x, ln1_g, ln1_b, act);

    // 3) fused QKV GEMM: [8192,3072] = xn @ WqkvT
    gemm_bf16_kernel<<<dim3(3072 / 128, 8192 / 128), 256, 0, stream>>>(
        act, wqkv, 8192, 3072, 1024, 0, nullptr, nullptr, nullptr, nullptr,
        qbuf, kbuf, vtb);

    // 4) flash attention -> cat (bf16)
    attn_kernel<<<dim3(64, 32), 128, 0, stream>>>(qbuf, kbuf, vtb, cat);

    // 5) O-proj + bo + residual x -> h (f32, stored in d_out)
    gemm_bf16_kernel<<<dim3(1024 / 128, 8192 / 128), 256, 0, stream>>>(
        cat, wot, 8192, 1024, 1024, 1, bo, x, out, nullptr,
        nullptr, nullptr, nullptr);

    // 6) LN2(h) -> hn (bf16, reuse act)
    ln_kernel<<<8192, 256, 0, stream>>>(out, ln2_g, ln2_b, act);

    // 7) FFN1: relu(hn @ W1 + b1) -> ff1 (bf16)
    gemm_bf16_kernel<<<dim3(4096 / 128, 8192 / 128), 256, 0, stream>>>(
        act, w1t, 8192, 4096, 1024, 2, b1, nullptr, nullptr, ff1,
        nullptr, nullptr, nullptr);

    // 8) FFN2: d_out = h + ff1 @ W2 + b2
    gemm_bf16_kernel<<<dim3(1024 / 128, 8192 / 128), 256, 0, stream>>>(
        ff1, w2t, 8192, 1024, 4096, 3, b2, nullptr, out, nullptr,
        nullptr, nullptr, nullptr);
}
